// seq2seqDecoder_38525856645428
// MI455X (gfx1250) — matrix operand order repaired and checked by reading, not compiled, not run on hardware
//
#include <hip/hip_runtime.h>

// Problem constants (from reference: B=256, T=96, F=32, H=1024)
#define B_   256
#define T_   96
#define F_   32
#define H_   1024
#define N4   4096          // 4*H
#define KTOT 1056          // H + F (fused h||x contraction dim)
#define KT_N 33            // KTOT / 32
#define NT_N 256           // N4 / 16
#define MT_N 16            // B_ / 16

typedef _Float16 half_t;
typedef __attribute__((ext_vector_type(16))) _Float16     v16h;
typedef __attribute__((ext_vector_type(8)))  float        v8f;
typedef __attribute__((ext_vector_type(4)))  unsigned int u32x4;
typedef __attribute__((ext_vector_type(8)))  int          i32x8;
typedef __attribute__((ext_vector_type(4)))  int          i32x4;

#if defined(__clang_major__) && (__clang_major__ >= 23)
#define TDM_LOAD(g0, g1, g2, g3) \
  __builtin_amdgcn_tensor_load_to_lds((g0), (g1), (g2), (g3), (i32x8){0,0,0,0,0,0,0,0}, 0)
#else
#define TDM_LOAD(g0, g1, g2, g3) \
  __builtin_amdgcn_tensor_load_to_lds((g0), (g1), (g2), (g3), 0)
#endif

// ---------------------------------------------------------------------------
// Fragment layouts (per CDNA5 ISA 7.12.2, wave32):
//  A 16x32 f16 tile: lane = (m&15) | (((k>>3)&1)<<4), slot = (k&7) | ((k&16)>>1)
//  B 32x16 f16 tile: lane = (n&15) | (((k>>3)&1)<<4), slot = (k&7) | ((k&16)>>1)
//  C/D 16x16 f32:    vgpr r holds (row = r + 8*(lane>>4), col = lane&15)
// Packed storage: [tile][lane][slot] contiguous -> one 32B vector load / lane.
// ---------------------------------------------------------------------------

__global__ __launch_bounds__(256)
void pack_weights_kernel(const float* __restrict__ W_x,
                         const float* __restrict__ W_h,
                         half_t* __restrict__ Wpack) {
  size_t p = (size_t)blockIdx.x * 256 + threadIdx.x;
  if (p >= (size_t)KTOT * N4) return;
  int slot = (int)(p & 15);
  int lane = (int)((p >> 4) & 31);
  size_t tmp = p >> 9;                 // tile index
  int nt = (int)(tmp & 255);
  int kt = (int)(tmp >> 8);
  int k = kt * 32 + ((lane >> 4) << 3) + (slot & 7) + ((slot >> 3) << 4);
  int n = nt * 16 + (lane & 15);
  float v = (k < H_) ? W_h[(size_t)k * N4 + n]
                     : W_x[(size_t)(k - H_) * N4 + n];
  Wpack[p] = (half_t)v;
}

__global__ __launch_bounds__(256)
void init_state_kernel(const float* __restrict__ initial_input,
                       const float* __restrict__ hidden_h,
                       const float* __restrict__ hidden_c,
                       float* __restrict__ c,
                       float* __restrict__ h,
                       half_t* __restrict__ Apack) {
  int idx = blockIdx.x * 256 + threadIdx.x;
  if (idx < B_ * H_) {
    c[idx] = hidden_c[idx];
    float hv = hidden_h[idx];
    h[idx] = hv;
    int b = idx >> 10, j = idx & 1023;
    int mt = b >> 4, m = b & 15, kt = j >> 5, sub = j & 31;
    int lane = m | (((sub >> 3) & 1) << 4);
    int slot = (sub & 7) | ((sub & 16) >> 1);
    Apack[((size_t)(mt * KT_N + kt) * 32 + lane) * 16 + slot] = (half_t)hv;
  } else if (idx < B_ * H_ + B_ * F_) {
    int i2 = idx - B_ * H_;
    int b = i2 >> 5, f = i2 & 31;
    float xv = initial_input[b * F_ + f];
    int lane = (b & 15) | (((f >> 3) & 1) << 4);
    int slot = (f & 7) | ((f & 16) >> 1);
    Apack[((size_t)((b >> 4) * KT_N + 32) * 32 + lane) * 16 + slot] = (half_t)xv;
  }
}

// TDM: move the two A tiles (mt0, mt0+1) for k-tile kt into LDS as one 2D
// tile: dim0 = 128 x 8B (one packed 16x32 f16 tile = 1KB), dim1 = 2 rows,
// row stride = KT_N*128 (8B units) = the Apack distance between m-tiles.
__device__ __forceinline__ void tdm_load_a_tiles(const half_t* Apack, int mt0,
                                                 int kt, unsigned lds_off) {
  unsigned long long ga = (unsigned long long)(uintptr_t)Apack +
                          (unsigned long long)(mt0 * KT_N + kt) * 1024ull;
  u32x4 g0;
  g0[0] = 1u;                                        // count=1 (valid D#)
  g0[1] = lds_off;                                   // lds_addr
  g0[2] = (unsigned)(ga & 0xFFFFFFFFull);            // global_addr[31:0]
  g0[3] = (unsigned)((ga >> 32) & 0x01FFFFFFull)     // global_addr[56:32]
        | (2u << 30);                                // type=2 ("image")
  i32x8 g1;
  g1[0] = (int)(3u << 16);                           // data_size=3 (8 bytes)
  g1[1] = (int)(128u << 16);                         // tensor_dim0 = 128
  g1[2] = 0;                                         // tensor_dim1.lo = 0
  g1[3] = (int)(1u | (128u << 16));                  // tensor_dim1=65536, tile_dim0=128
  g1[4] = 2;                                         // tile_dim1 = 2
  g1[5] = (int)(KT_N * 128);                         // tensor_dim0_stride = 4224
  g1[6] = 0;
  g1[7] = 0;
  i32x4 z4 = {0, 0, 0, 0};
  TDM_LOAD(g0, g1, z4, z4);
}

// z[256,4096] = [h||x](256x1056, f16) @ [W_h;W_x](1056x4096, f16), fp32 acc.
// WG = 8 waves. Wave computes a 32x32 output block (4 accumulators).
// WG covers 32 rows x 256 cols. Grid = (16, 8). A tiles arrive via TDM into
// double-buffered LDS; wave0 drives the DMA pipeline.
__global__ __launch_bounds__(256)
void lstm_gemm_kernel(const half_t* __restrict__ Apack,
                      const half_t* __restrict__ Wpack,
                      float* __restrict__ z) {
  __shared__ __align__(32) half_t lds[2048];   // 2 buffers x (two 16x32 tiles)
  const int tid  = threadIdx.x;
  const int lane = tid & 31;
  const int w    = tid >> 5;
  const int mt0  = blockIdx.y * 2;
  const int nt0  = blockIdx.x * 16 + w * 2;
  const unsigned lds_base = (unsigned)(uintptr_t)(&lds[0]);

  v8f acc00 = {}; v8f acc01 = {}; v8f acc10 = {}; v8f acc11 = {};

  if (tid < 32) {                               // wave0: prime the pipeline
    tdm_load_a_tiles(Apack, mt0, 0, lds_base);
  }

  for (int kt = 0; kt < KT_N; ++kt) {
    const int cur = kt & 1;
    if (tid < 32) {
      if (kt + 1 < KT_N) {
        tdm_load_a_tiles(Apack, mt0, kt + 1, lds_base + (unsigned)((cur ^ 1) * 2048));
        __builtin_amdgcn_s_wait_tensorcnt(1);   // buffer `cur` has landed
      } else {
        __builtin_amdgcn_s_wait_tensorcnt(0);
      }
    }
    __syncthreads();                            // A tiles visible to all waves

    const half_t* abuf = lds + cur * 1024;
    v16h a0 = *(const v16h*)(abuf + lane * 16);
    v16h a1 = *(const v16h*)(abuf + 512 + lane * 16);

    const half_t* bbase = Wpack + (size_t)kt * NT_N * 512;
    v16h b0 = *(const v16h*)(bbase + (size_t)nt0 * 512 + lane * 16);
    v16h b1 = *(const v16h*)(bbase + (size_t)(nt0 + 1) * 512 + lane * 16);

    if (kt + 1 < KT_N) {   // pull next k-slab of B toward the WGP (L2-resident)
      __builtin_prefetch(Wpack + ((size_t)(kt + 1) * NT_N + nt0) * 512 + lane * 16, 0, 1);
    }

    acc00 = __builtin_amdgcn_wmma_f32_16x16x32_f16(false, a0, false, b0, (short)0, acc00, false, false);
    acc01 = __builtin_amdgcn_wmma_f32_16x16x32_f16(false, a0, false, b1, (short)0, acc01, false, false);
    acc10 = __builtin_amdgcn_wmma_f32_16x16x32_f16(false, a1, false, b0, (short)0, acc10, false, false);
    acc11 = __builtin_amdgcn_wmma_f32_16x16x32_f16(false, a1, false, b1, (short)0, acc11, false, false);
    __syncthreads();       // all waves done with buffer `cur` before reuse
  }

  const int col    = lane & 15;
  const int rowOff = (lane >> 4) * 8;
#pragma unroll
  for (int r = 0; r < 8; ++r) {
    int row0 = mt0 * 16 + rowOff + r;
    int row1 = row0 + 16;
    z[(size_t)row0 * N4 + (size_t)nt0 * 16 + col]       = acc00[r];
    z[(size_t)row0 * N4 + (size_t)(nt0 + 1) * 16 + col] = acc01[r];
    z[(size_t)row1 * N4 + (size_t)nt0 * 16 + col]       = acc10[r];
    z[(size_t)row1 * N4 + (size_t)(nt0 + 1) * 16 + col] = acc11[r];
  }
}

// Gates + state update + re-pack h into A fragments for the next step.
__global__ __launch_bounds__(256)
void lstm_gate_kernel(const float* __restrict__ z,
                      const float* __restrict__ bias,
                      float* __restrict__ c,
                      float* __restrict__ h,
                      half_t* __restrict__ Apack) {
  int idx = blockIdx.x * 256 + threadIdx.x;    // 0 .. B*H-1
  int b = idx >> 10;
  int j = idx & 1023;
  const float* zr = z + (size_t)b * N4;
  float zi = zr[j]        + bias[j];
  float zf = zr[j + 1024] + bias[j + 1024];
  float zg = zr[j + 2048] + bias[j + 2048];
  float zo = zr[j + 3072] + bias[j + 3072];
  float ig = 1.0f / (1.0f + __expf(-zi));
  float fg = 1.0f / (1.0f + __expf(-zf));
  float gg = tanhf(zg);
  float og = 1.0f / (1.0f + __expf(-zo));
  float cn = fg * c[idx] + ig * gg;
  c[idx] = cn;
  float hn = og * tanhf(cn);
  h[idx] = hn;
  int mt = b >> 4, m = b & 15, kt = j >> 5, sub = j & 31;
  int lane = m | (((sub >> 3) & 1) << 4);
  int slot = (sub & 7) | ((sub & 16) >> 1);
  Apack[((size_t)(mt * KT_N + kt) * 32 + lane) * 16 + slot] = (half_t)hn;
}

// y = h @ W_dense + b_dense; write output and feed back as next-step x frags.
__global__ __launch_bounds__(256)
void lstm_dense_kernel(const float* __restrict__ h,
                       const float* __restrict__ Wd,
                       const float* __restrict__ bd,
                       float* __restrict__ out,
                       half_t* __restrict__ Apack,
                       int t) {
  int idx = blockIdx.x * 256 + threadIdx.x;    // 0 .. B*F-1
  int b = idx >> 5;
  int f = idx & 31;
  const float* hr = h + (size_t)b * H_;
  float sum = bd[f];
#pragma unroll 4
  for (int k = 0; k < H_; ++k) sum = fmaf(hr[k], Wd[k * F_ + f], sum);
  out[((size_t)b * T_ + t) * F_ + f] = sum;
  int lane = (b & 15) | (((f >> 3) & 1) << 4);
  int slot = (f & 7) | ((f & 16) >> 1);
  Apack[((size_t)((b >> 4) * KT_N + 32) * 32 + lane) * 16 + slot] = (half_t)sum;
}

extern "C" void kernel_launch(void* const* d_in, const int* in_sizes, int n_in,
                              void* d_out, int out_size, void* d_ws, size_t ws_size,
                              hipStream_t stream) {
  const float* initial_input = (const float*)d_in[0];
  const float* hidden_h      = (const float*)d_in[1];
  const float* hidden_c      = (const float*)d_in[2];
  // d_in[3] = targets (only defines T, unused numerically)
  const float* W_x     = (const float*)d_in[4];
  const float* W_h     = (const float*)d_in[5];
  const float* bias    = (const float*)d_in[6];
  const float* W_dense = (const float*)d_in[7];
  const float* b_dense = (const float*)d_in[8];
  float* out = (float*)d_out;

  char* ws = (char*)d_ws;
  size_t off = 0;
  half_t* Wpack = (half_t*)(ws + off); off += (size_t)KTOT * N4 * sizeof(half_t);          // 8.65 MB
  half_t* Apack = (half_t*)(ws + off); off += (size_t)MT_N * KT_N * 512 * sizeof(half_t);  // 0.54 MB
  float*  z     = (float*)(ws + off);  off += (size_t)B_ * N4 * sizeof(float);             // 4.0 MB
  float*  c     = (float*)(ws + off);  off += (size_t)B_ * H_ * sizeof(float);             // 1.0 MB
  float*  h     = (float*)(ws + off);  off += (size_t)B_ * H_ * sizeof(float);             // 1.0 MB

  pack_weights_kernel<<<(KTOT * N4 + 255) / 256, 256, 0, stream>>>(W_x, W_h, Wpack);
  init_state_kernel<<<(B_ * H_ + B_ * F_ + 255) / 256, 256, 0, stream>>>(
      initial_input, hidden_h, hidden_c, c, h, Apack);

  dim3 gemmGrid(N4 / 256, B_ / 32);   // (16, 8)
  for (int t = 0; t < T_; ++t) {
    lstm_gemm_kernel<<<gemmGrid, 256, 0, stream>>>(Apack, Wpack, z);
    lstm_gate_kernel<<<(B_ * H_) / 256, 256, 0, stream>>>(z, bias, c, h, Apack);
    lstm_dense_kernel<<<(B_ * F_) / 256, 256, 0, stream>>>(h, W_dense, b_dense, out, Apack, t);
  }
}
